// MyModel_87522843561434
// MI455X (gfx1250) — matrix operand order repaired and checked by reading, not compiled, not run on hardware
//
#include <hip/hip_runtime.h>
#include <hip/hip_bf16.h>
#include <math.h>

typedef __attribute__((ext_vector_type(16))) _Float16 v16h;
typedef __attribute__((ext_vector_type(8)))  float    v8f;
typedef __attribute__((__vector_size__(4 * sizeof(int)))) int v4i_vs;

#define EPS 1e-3f
#define B_TOT 4096
#define T_IN  512
#define F_IN  10
#define T1    510   // after conv1 (K=3)
#define T2    505   // after conv2 (K=6)
#define TS    512   // padded T stride in workspace

// Fast transcendental gate math (v_exp_f32 TRANS path, saturation-safe)
__device__ __forceinline__ float sigf(float x)
{
    return 1.0f / (1.0f + __expf(-x));
}
__device__ __forceinline__ float tanh_fast(float x)
{
    // 2*sigmoid(2x) - 1 ; e->inf gives -1, e->0 gives +1 (no NaN at extremes)
    return 2.0f / (1.0f + __expf(-2.0f * x)) - 1.0f;
}

// --- gfx1250 async global->LDS copy (ASYNCcnt), with safe fallbacks ---------
__device__ __forceinline__ void async_copy_b128(const void* gsrc, void* ldst)
{
#if __has_builtin(__builtin_amdgcn_global_load_async_to_lds_b128)
    __builtin_amdgcn_global_load_async_to_lds_b128(
        (__attribute__((address_space(1))) v4i_vs*)(size_t)gsrc,
        (__attribute__((address_space(3))) v4i_vs*)(unsigned)(size_t)ldst,
        0, 0);
#else
    *(uint4*)ldst = *(const uint4*)gsrc;   // synchronous fallback
#endif
}
__device__ __forceinline__ void wait_async()
{
#if __has_builtin(__builtin_amdgcn_s_wait_asynccnt)
    __builtin_amdgcn_s_wait_asynccnt(0);
#else
    asm volatile("s_wait_asynccnt 0" ::: "memory");
#endif
}

// ---------------------------------------------------------------------------
// Kernel 1a: BN1 -> conv1d(K=3, 10->16) -> ReLU -> BN2 ; output f16 [B][TS][16]
// ---------------------------------------------------------------------------
__global__ __launch_bounds__(256) void conv1_kernel(
    const float* __restrict__ x,
    const float* __restrict__ g1, const float* __restrict__ b1,
    const float* __restrict__ m1, const float* __restrict__ v1,
    const float* __restrict__ w1, const float* __restrict__ cb1,
    const float* __restrict__ g2, const float* __restrict__ b2,
    const float* __restrict__ m2, const float* __restrict__ v2,
    _Float16* __restrict__ h1)
{
    int gid = blockIdx.x * blockDim.x + threadIdx.x;
    if (gid >= B_TOT * T1) return;
    int b = gid / T1, t = gid % T1;

    float xn[3][10];
#pragma unroll
    for (int k = 0; k < 3; ++k)
#pragma unroll
        for (int f = 0; f < F_IN; ++f) {
            float xv = x[((size_t)b * T_IN + (t + k)) * F_IN + f];
            xn[k][f] = (xv - m1[f]) * (g1[f] * rsqrtf(v1[f] + EPS)) + b1[f];
        }

    float acc[16];
#pragma unroll
    for (int c = 0; c < 16; ++c) acc[c] = cb1[c];
#pragma unroll
    for (int k = 0; k < 3; ++k)
#pragma unroll
        for (int f = 0; f < F_IN; ++f) {
            float xv = xn[k][f];
            const float* wr = &w1[(k * F_IN + f) * 16];
#pragma unroll
            for (int c = 0; c < 16; ++c) acc[c] = fmaf(xv, wr[c], acc[c]);
        }

    _Float16* dst = &h1[((size_t)b * TS + t) * 16];
#pragma unroll
    for (int c = 0; c < 16; ++c) {
        float y = fmaxf(acc[c], 0.0f);
        y = (y - m2[c]) * (g2[c] * rsqrtf(v2[c] + EPS)) + b2[c];
        dst[c] = (_Float16)y;
    }
}

// ---------------------------------------------------------------------------
// Kernel 1b: conv1d(K=6, 16->32) -> ReLU -> BN3 ; output f16 [B][TS][32]
// ---------------------------------------------------------------------------
__global__ __launch_bounds__(256) void conv2_kernel(
    const _Float16* __restrict__ h1,
    const float* __restrict__ w2, const float* __restrict__ cb2,
    const float* __restrict__ g3, const float* __restrict__ b3,
    const float* __restrict__ m3, const float* __restrict__ v3,
    _Float16* __restrict__ h2)
{
    int gid = blockIdx.x * blockDim.x + threadIdx.x;
    if (gid >= B_TOT * T2) return;
    int b = gid / T2, t = gid % T2;

    float acc[32];
#pragma unroll
    for (int c = 0; c < 32; ++c) acc[c] = cb2[c];

    for (int k = 0; k < 6; ++k) {
        const _Float16* row = &h1[((size_t)b * TS + (t + k)) * 16];
        float rf[16];
#pragma unroll
        for (int f = 0; f < 16; ++f) rf[f] = (float)row[f];
#pragma unroll
        for (int f = 0; f < 16; ++f) {
            const float* wr = &w2[(k * 16 + f) * 32];
            float xv = rf[f];
#pragma unroll
            for (int c = 0; c < 32; ++c) acc[c] = fmaf(xv, wr[c], acc[c]);
        }
    }

    _Float16* dst = &h2[((size_t)b * TS + t) * 32];
#pragma unroll
    for (int c = 0; c < 32; ++c) {
        float y = fmaxf(acc[c], 0.0f);
        y = (y - m3[c]) * (g3[c] * rsqrtf(v3[c] + EPS)) + b3[c];
        dst[c] = (_Float16)y;
    }
}

// ---------------------------------------------------------------------------
// Fused 3-layer LSTM + Dense + softmax. One wave (32 lanes) per 16 batch rows.
// WMMA 16x16x32 f16->f32; weights packed in LDS B-layout; states in LDS.
// ---------------------------------------------------------------------------

// A-operand (16x32 f16): lane -> M=lane&15, half=lane>>4;
// VGPR v holds K pair k0 = 2*(v&3) + 16*(v>>2) + 8*half  (ISA 7.12.2 table)
__device__ __forceinline__ v16h load_A(const _Float16* __restrict__ buf, int lane)
{
    int m = lane & 15, half = lane >> 4;
    v16h a;
#pragma unroll
    for (int v = 0; v < 8; ++v) {
        int k0 = 2 * (v & 3) + 16 * (v >> 2) + 8 * half;
        a[2 * v]     = buf[m * 32 + k0];
        a[2 * v + 1] = buf[m * 32 + k0 + 1];
    }
    return a;
}

// Pack row-major [Kdim][Ndim] f32 weights into LDS B-layout tiles (K padded to
// 32 with zeros). B 32x16: lane -> n=lane&15; VGPR v holds K pair
// k = 2*(v&3) + 16*(v>>2) + 8*(lane>>4) (+0/1). 16 f16 per lane, contiguous per (tile,lane).
__device__ void pack_B(const float* __restrict__ G, int Kdim, int Ndim,
                       _Float16* __restrict__ dst, int lane)
{
    int ntiles = Ndim / 16;
    int n0 = lane & 15, half = lane >> 4;
    for (int nt = 0; nt < ntiles; ++nt) {
        int n = n0 + nt * 16;
#pragma unroll
        for (int v = 0; v < 8; ++v)
#pragma unroll
            for (int j = 0; j < 2; ++j) {
                int k = 2 * (v & 3) + 16 * (v >> 2) + 8 * half + j;
                float val = (k < Kdim) ? G[k * Ndim + n] : 0.0f;
                dst[(nt * 32 + lane) * 16 + 2 * v + j] = (_Float16)val;
            }
    }
}

__device__ __forceinline__ v8f wmma_f16(v16h a, v16h b, v8f c)
{
    return __builtin_amdgcn_wmma_f32_16x16x32_f16(false, a, false, b, (short)0, c,
                                                  false, false);
}

// One LSTM timestep for a 16-row batch tile.
__device__ void lstm_step(int lane,
                          const _Float16* __restrict__ xA,
                          _Float16* __restrict__ hbuf,
                          float* __restrict__ cbuf,
                          float* __restrict__ zbuf,
                          const _Float16* __restrict__ sW,
                          const _Float16* __restrict__ sU,
                          const float* __restrict__ sb,
                          int H)
{
    const int H4 = 4 * H;
    const int ntiles = H4 / 16;
    const int nlo = lane & 15, half = lane >> 4;

    v16h xa = load_A(xA, lane);
    v16h ha = load_A(hbuf, lane);

    for (int nt = 0; nt < ntiles; ++nt) {
        v16h bw = *(const v16h*)&sW[(nt * 32 + lane) * 16];
        v16h bu = *(const v16h*)&sU[(nt * 32 + lane) * 16];
        v8f acc = {};
        acc = wmma_f16(xa, bw, acc);   // z  = x_t @ W
        acc = wmma_f16(ha, bu, acc);   // z += h_{t-1} @ U
        int n = nlo + nt * 16;
#pragma unroll
        for (int r = 0; r < 8; ++r)
            zbuf[(r + 8 * half) * H4 + n] = acc[r];   // C/D layout -> [M][N]
    }
    __syncthreads();

    for (int e = lane; e < 16 * H; e += 32) {
        int m = e / H, j = e % H;
        float zi = zbuf[m * H4 + j]          + sb[j];
        float zf = zbuf[m * H4 + H + j]      + sb[H + j];
        float zg = zbuf[m * H4 + 2 * H + j]  + sb[2 * H + j];
        float zo = zbuf[m * H4 + 3 * H + j]  + sb[3 * H + j];
        float c = sigf(zf) * cbuf[m * H + j] + sigf(zi) * tanh_fast(zg);
        cbuf[m * H + j] = c;
        hbuf[m * 32 + j] = (_Float16)(sigf(zo) * tanh_fast(c));
    }
    __syncthreads();
}

__global__ __launch_bounds__(32) void lstm_dense_kernel(
    const _Float16* __restrict__ h2,
    const float* __restrict__ l1W, const float* __restrict__ l1U, const float* __restrict__ l1b,
    const float* __restrict__ l2W, const float* __restrict__ l2U, const float* __restrict__ l2b,
    const float* __restrict__ l3W, const float* __restrict__ l3U, const float* __restrict__ l3b,
    const float* __restrict__ d1w, const float* __restrict__ d1b,
    const float* __restrict__ ow,  const float* __restrict__ ob,
    float* __restrict__ out)
{
    __shared__ __align__(32) _Float16 sW1[8 * 32 * 16], sU1[8 * 32 * 16];
    __shared__ __align__(32) _Float16 sW2[6 * 32 * 16], sU2[6 * 32 * 16];
    __shared__ __align__(32) _Float16 sW3[4 * 32 * 16], sU3[4 * 32 * 16];
    __shared__ float sB1[128], sB2[96], sB3[64];
    __shared__ __align__(32) float    zbuf[16 * 128];
    __shared__ __align__(32) _Float16 xbuf[2][16 * 32];   // double-buffered
    __shared__ __align__(32) _Float16 hbuf1[16 * 32], hbuf2[16 * 32], hbuf3[16 * 32];
    __shared__ float cbuf1[16 * 32], cbuf2[16 * 24], cbuf3[16 * 16];

    const int lane = threadIdx.x;
    const int b0 = blockIdx.x * 16;

    pack_B(l1W, 32, 128, sW1, lane); pack_B(l1U, 32, 128, sU1, lane);
    pack_B(l2W, 32,  96, sW2, lane); pack_B(l2U, 24,  96, sU2, lane);
    pack_B(l3W, 24,  64, sW3, lane); pack_B(l3U, 16,  64, sU3, lane);
    for (int i = lane; i < 128; i += 32) sB1[i] = l1b[i];
    for (int i = lane; i <  96; i += 32) sB2[i] = l2b[i];
    for (int i = lane; i <  64; i += 32) sB3[i] = l3b[i];
    for (int i = lane; i < 16 * 32; i += 32) {
        hbuf1[i] = (_Float16)0.0f; hbuf2[i] = (_Float16)0.0f; hbuf3[i] = (_Float16)0.0f;
        cbuf1[i] = 0.0f;
    }
    for (int i = lane; i < 16 * 24; i += 32) cbuf2[i] = 0.0f;
    for (int i = lane; i < 16 * 16; i += 32) cbuf3[i] = 0.0f;
    __syncthreads();

    // Per-lane slice of a 16x32 f16 tile: 2 x 16B chunks.
    const int row = lane >> 1, chunk = lane & 1;
    const int e0 = (2 * chunk) * 8;     // f16 element offset of first 16B chunk
    const int e1 = (2 * chunk + 1) * 8;

    // Preload t = 0 via async-to-LDS, overlapped thereafter.
    {
        const _Float16* g = h2 + ((size_t)(b0 + row) * TS + 0) * 32;
        async_copy_b128(g + e0, &xbuf[0][row * 32 + e0]);
        async_copy_b128(g + e1, &xbuf[0][row * 32 + e1]);
        wait_async();
    }

    for (int t = 0; t < T2; ++t) {
        const int cur = t & 1;
        if (t + 1 < T2) {   // issue next tile before compute (overlap with WMMA)
            const _Float16* g = h2 + ((size_t)(b0 + row) * TS + (t + 1)) * 32;
            async_copy_b128(g + e0, &xbuf[1 - cur][row * 32 + e0]);
            async_copy_b128(g + e1, &xbuf[1 - cur][row * 32 + e1]);
        }
        if (t + 16 < T2)    // deep stream-ahead hint -> global_prefetch_b8
            __builtin_prefetch(h2 + ((size_t)(b0 + row) * TS + (t + 16)) * 32, 0, 3);
        __syncthreads();

        lstm_step(lane, xbuf[cur], hbuf1, cbuf1, zbuf, sW1, sU1, sB1, 32);
        lstm_step(lane, hbuf1,     hbuf2, cbuf2, zbuf, sW2, sU2, sB2, 24);
        lstm_step(lane, hbuf2,     hbuf3, cbuf3, zbuf, sW3, sU3, sB3, 16);

        wait_async();       // next x tile resident before next iteration
    }

    // Dense(16->8)+ReLU, Dense(8->5), softmax — one lane per batch row.
    if (lane < 16) {
        int m = lane;
        float hh[16];
#pragma unroll
        for (int k = 0; k < 16; ++k) hh[k] = (float)hbuf3[m * 32 + k];
        float d1[8];
#pragma unroll
        for (int j = 0; j < 8; ++j) {
            float a = d1b[j];
#pragma unroll
            for (int k = 0; k < 16; ++k) a = fmaf(hh[k], d1w[k * 8 + j], a);
            d1[j] = fmaxf(a, 0.0f);
        }
        float lg[5];
        float mx = -1e30f;
#pragma unroll
        for (int j = 0; j < 5; ++j) {
            float a = ob[j];
#pragma unroll
            for (int k = 0; k < 8; ++k) a = fmaf(d1[k], ow[k * 5 + j], a);
            lg[j] = a;
            mx = fmaxf(mx, a);
        }
        float s = 0.0f;
#pragma unroll
        for (int j = 0; j < 5; ++j) { lg[j] = __expf(lg[j] - mx); s += lg[j]; }
        float inv = 1.0f / s;
#pragma unroll
        for (int j = 0; j < 5; ++j) out[(size_t)(b0 + m) * 5 + j] = lg[j] * inv;
    }
}

// ---------------------------------------------------------------------------
extern "C" void kernel_launch(void* const* d_in, const int* in_sizes, int n_in,
                              void* d_out, int out_size, void* d_ws, size_t ws_size,
                              hipStream_t stream)
{
    const float* x    = (const float*)d_in[0];
    const float* g1   = (const float*)d_in[1];
    const float* b1   = (const float*)d_in[2];
    const float* m1   = (const float*)d_in[3];
    const float* v1   = (const float*)d_in[4];
    const float* g2   = (const float*)d_in[5];
    const float* b2   = (const float*)d_in[6];
    const float* m2   = (const float*)d_in[7];
    const float* v2   = (const float*)d_in[8];
    const float* g3   = (const float*)d_in[9];
    const float* b3   = (const float*)d_in[10];
    const float* m3   = (const float*)d_in[11];
    const float* v3   = (const float*)d_in[12];
    const float* w1   = (const float*)d_in[13];
    const float* cb1  = (const float*)d_in[14];
    const float* w2   = (const float*)d_in[15];
    const float* cb2  = (const float*)d_in[16];
    const float* l1W  = (const float*)d_in[17];
    const float* l1U  = (const float*)d_in[18];
    const float* l1b  = (const float*)d_in[19];
    const float* l2W  = (const float*)d_in[20];
    const float* l2U  = (const float*)d_in[21];
    const float* l2b  = (const float*)d_in[22];
    const float* l3W  = (const float*)d_in[23];
    const float* l3U  = (const float*)d_in[24];
    const float* l3b  = (const float*)d_in[25];
    const float* d1w  = (const float*)d_in[26];
    const float* d1b  = (const float*)d_in[27];
    const float* ow   = (const float*)d_in[28];
    const float* ob   = (const float*)d_in[29];
    float* out = (float*)d_out;

    _Float16* h1 = (_Float16*)d_ws;
    _Float16* h2 = h1 + (size_t)B_TOT * TS * 16;

    {
        int n = B_TOT * T1;
        conv1_kernel<<<(n + 255) / 256, 256, 0, stream>>>(
            x, g1, b1, m1, v1, w1, cb1, g2, b2, m2, v2, h1);
    }
    {
        int n = B_TOT * T2;
        conv2_kernel<<<(n + 255) / 256, 256, 0, stream>>>(
            h1, w2, cb2, g3, b3, m3, v3, h2);
    }
    {
        lstm_dense_kernel<<<B_TOT / 16, 32, 0, stream>>>(
            h2, l1W, l1U, l1b, l2W, l2U, l2b, l3W, l3U, l3b,
            d1w, d1b, ow, ob, out);
    }
}
